// GatedRecurrentAttention_13288628814319
// MI455X (gfx1250) — compile-verified, matrix operand order checked by reading, not run on hardware
//
#include <hip/hip_runtime.h>
#include <hip/hip_bf16.h>

// ---------------------------------------------------------------------------
// Types / constants
// ---------------------------------------------------------------------------
typedef __attribute__((ext_vector_type(16))) __bf16 v16bf;
typedef __attribute__((ext_vector_type(8)))  __bf16 bf16x8;
typedef __attribute__((ext_vector_type(8)))  float  v8f;
typedef __attribute__((ext_vector_type(4)))  unsigned int u32x4;
typedef __attribute__((ext_vector_type(8)))  int    i32x8;
typedef __attribute__((ext_vector_type(4)))  int    i32x4;

#define GB 8        // batch
#define GT 1024     // seq len
#define GD 1024     // model dim
#define GH 16       // heads
#define GDH 64      // head dim
#define NSTEPS 4
#define SCAN_NWG 64
#define SCAN_THREADS 128

#if defined(__gfx1250__) && defined(__has_builtin)
#  if __has_builtin(__builtin_amdgcn_tensor_load_to_lds) && __has_builtin(__builtin_amdgcn_s_wait_tensorcnt)
#    define HAVE_TDM 1
#  endif
#endif
#ifndef HAVE_TDM
#  define HAVE_TDM 0
#endif

#define WMMA_BF16(a, b, c) \
  __builtin_amdgcn_wmma_f32_16x16x32_bf16(false, (a), false, (b), (short)0, (c), false, false)

// ---------------------------------------------------------------------------
// Fragment helpers (ISA 7.12.2 layouts, wave32)
// A 16x32 bf16: lanes 0-15 row M=lane, K 0..7 + 16..23; lanes 16-31: +8 on K.
// B 32x16 (row-major [N][K], i.e. weight rows are outputs): mirrored layout.
// Both reduce to two contiguous 8-element (16B) chunks per lane.
// ---------------------------------------------------------------------------
__device__ inline v16bf load_frag16(const __bf16* tile, int ld, int lane) {
  const int r   = lane & 15;
  const int khi = (lane >> 4) & 1;
  const __bf16* p = tile + (size_t)r * ld + 8 * khi;
  union { bf16x8 h[2]; v16bf v; } u;
  u.h[0] = *(const bf16x8*)(p);
  u.h[1] = *(const bf16x8*)(p + 16);
  return u.v;
}

// B fragment from a [K][N] (k-major) tile, e.g. V in attention.
__device__ inline v16bf load_frag16_kmajor(const __bf16* tile, int ld, int lane) {
  const int n   = lane & 15;
  const int khi = (lane >> 4) & 1;
  v16bf f{};
#pragma unroll
  for (int e = 0; e < 16; ++e) {
    int v = e >> 1;
    int k = ((v < 4) ? (2 * v) : (16 + 2 * (v - 4))) + (e & 1) + 8 * khi;
    f[e] = tile[(size_t)k * ld + n];
  }
  return f;
}

// C/D 16x16 f32: lanes 0-15 rows 0..7 (vgpr r), lanes 16-31 rows 8..15, col=lane&15
__device__ inline void store_c_tile(float* dst, int ld, int lane, v8f c) {
  const int col = lane & 15;
  const int rb  = (lane >> 4) * 8;
#pragma unroll
  for (int r = 0; r < 8; ++r) dst[(size_t)(rb + r) * ld + col] = c[r];
}

// ---------------------------------------------------------------------------
// fp32 -> bf16 cast
// ---------------------------------------------------------------------------
__global__ void cast_f32_bf16(const float* __restrict__ s, __bf16* __restrict__ d, int n) {
  int i = blockIdx.x * blockDim.x + threadIdx.x;
  if (i < n) d[i] = (__bf16)s[i];
}

// zero h state + grid-barrier counters (re-run every launch for determinism)
__global__ void init_state(__bf16* h16, unsigned* bar) {
  int i = blockIdx.x * blockDim.x + threadIdx.x;
  if (i < 16 * GD) h16[i] = (__bf16)0.f;
  if (i < 2) bar[i] = 0u;
}

// ---------------------------------------------------------------------------
// C[M,N] = A[M,K] @ W[N,K]^T + bias ;  A,W bf16 row-major; out f32 and/or bf16
// 128x128 tile per WG, 256 threads = 8 waves (2 M x 4 N), wave tile 64x32.
// M % 128 == 0, N % 128 == 0, K % 32 == 0 (true for all uses here).
// ---------------------------------------------------------------------------
__global__ __launch_bounds__(256)
void gemm_bf16_nt(const __bf16* __restrict__ A, const __bf16* __restrict__ W,
                  const float* __restrict__ bias, float* __restrict__ Cf,
                  __bf16* __restrict__ Cb, int M, int N, int K) {
  __shared__ __bf16 sA[128 * 32];
  __shared__ __bf16 sW[128 * 32];
  const int tid = threadIdx.x, lane = tid & 31, wave = tid >> 5;
  const int bm = blockIdx.x, bn = blockIdx.y;
  const int wm = (wave & 1) * 64;
  const int wn = (wave >> 1) * 32;

  v8f acc[4][2] = {};

  const int row  = tid >> 1;          // 0..127
  const int colb = (tid & 1) * 16;    // 0 / 16
  const __bf16* gA = A + (size_t)(bm * 128 + row) * K + colb;
  const __bf16* gW = W + (size_t)(bn * 128 + row) * K + colb;
  __bf16* lA = sA + row * 32 + colb;
  __bf16* lW = sW + row * 32 + colb;

  for (int k0 = 0; k0 < K; k0 += 32) {
    uint4 a0 = *(const uint4*)(gA + k0);
    uint4 a1 = *(const uint4*)(gA + k0 + 8);
    uint4 w0 = *(const uint4*)(gW + k0);
    uint4 w1 = *(const uint4*)(gW + k0 + 8);
    if (lane == 0 && k0 + 64 < K) {         // warm next-next tile in L2/L0
      __builtin_prefetch(gA + k0 + 64, 0, 1);
      __builtin_prefetch(gW + k0 + 64, 0, 1);
    }
    *(uint4*)lA = a0; *(uint4*)(lA + 8) = a1;
    *(uint4*)lW = w0; *(uint4*)(lW + 8) = w1;
    __syncthreads();
#pragma unroll
    for (int j = 0; j < 2; ++j) {
      v16bf bf = load_frag16(sW + (wn + j * 16) * 32, 32, lane);
#pragma unroll
      for (int i = 0; i < 4; ++i) {
        v16bf af = load_frag16(sA + (wm + i * 16) * 32, 32, lane);
        acc[i][j] = WMMA_BF16(af, bf, acc[i][j]);
      }
    }
    __syncthreads();
  }

  const int ccol = lane & 15;
  const int rb   = (lane >> 4) * 8;
#pragma unroll
  for (int j = 0; j < 2; ++j) {
    const int gcol = bn * 128 + wn + j * 16 + ccol;
    const float bv = bias ? bias[gcol] : 0.f;
#pragma unroll
    for (int i = 0; i < 4; ++i) {
      const int grow0 = bm * 128 + wm + i * 16 + rb;
#pragma unroll
      for (int r = 0; r < 8; ++r) {
        float v = acc[i][j][r] + bv;
        size_t idx = (size_t)(grow0 + r) * N + gcol;
        if (Cf) Cf[idx] = v;
        if (Cb) Cb[idx] = (__bf16)v;
      }
    }
  }
}

// ---------------------------------------------------------------------------
// Flash attention: grid (B*H, T/64), 128 threads = 4 waves; each wave owns a
// 16-row Q tile (Dh=64), streams 64-row K/V tiles, online softmax in f32.
// qkv layout: [B*T, 3D] bf16 (q | k | v), head h at column h*64 within each D.
// ---------------------------------------------------------------------------
__global__ __launch_bounds__(128)
void flash_attn(const __bf16* __restrict__ qkv, __bf16* __restrict__ ctx) {
  __shared__ __bf16 sK[64 * 64];
  __shared__ __bf16 sV[64 * 64];
  __shared__ __bf16 sP[4][16 * 64];
  const int tid = threadIdx.x, lane = tid & 31, wave = tid >> 5;
  const int b = blockIdx.x / GH, h = blockIdx.x % GH;
  const int q0 = blockIdx.y * 64;
  const int ld = 3 * GD;
  const size_t base = (size_t)b * GT * ld;
  const __bf16* Qp = qkv + base + (size_t)(q0 + wave * 16) * ld + h * GDH;
  const __bf16* Kp = qkv + base + GD + h * GDH;
  const __bf16* Vp = qkv + base + 2 * GD + h * GDH;

  // Q fragments (K-dim 64 -> 2 fragments), pre-scaled by 1/sqrt(Dh)
  const float scale = 0.125f;
  v16bf qf[2];
#pragma unroll
  for (int kk = 0; kk < 2; ++kk) {
    v16bf f = load_frag16(Qp + kk * 32, ld, lane);
#pragma unroll
    for (int e = 0; e < 16; ++e) f[e] = (__bf16)((float)f[e] * scale);
    qf[kk] = f;
  }

  v8f o[4] = {};
  float m[8], l[8];
#pragma unroll
  for (int r = 0; r < 8; ++r) { m[r] = -1e30f; l[r] = 0.f; }

  const int crow = tid >> 1, chalf = (tid & 1) * 32;   // cooperative tile load
  for (int kt = 0; kt < GT; kt += 64) {
    const __bf16* kg = Kp + (size_t)(kt + crow) * ld + chalf;
    const __bf16* vg = Vp + (size_t)(kt + crow) * ld + chalf;
#pragma unroll
    for (int q = 0; q < 4; ++q) {
      *(uint4*)(sK + crow * 64 + chalf + q * 8) = *(const uint4*)(kg + q * 8);
      *(uint4*)(sV + crow * 64 + chalf + q * 8) = *(const uint4*)(vg + q * 8);
    }
    __syncthreads();

    // S = Q K^T  (16 x 64)
    v8f s[4] = {};
#pragma unroll
    for (int kk = 0; kk < 2; ++kk)
#pragma unroll
      for (int n = 0; n < 4; ++n) {
        v16bf bf = load_frag16(sK + (n * 16) * 64 + kk * 32, 64, lane);
        s[n] = WMMA_BF16(qf[kk], bf, s[n]);
      }

    // online softmax; row data lives in 16-lane groups (width-16 shuffles)
    float alpha[8];
#pragma unroll
    for (int r = 0; r < 8; ++r) {
      float mx = fmaxf(fmaxf(s[0][r], s[1][r]), fmaxf(s[2][r], s[3][r]));
      mx = fmaxf(mx, __shfl_xor(mx, 1, 16));
      mx = fmaxf(mx, __shfl_xor(mx, 2, 16));
      mx = fmaxf(mx, __shfl_xor(mx, 4, 16));
      mx = fmaxf(mx, __shfl_xor(mx, 8, 16));
      float mnew = fmaxf(m[r], mx);
      alpha[r] = __expf(m[r] - mnew);
      m[r] = mnew;
      float rs = 0.f;
#pragma unroll
      for (int n = 0; n < 4; ++n) { float e = __expf(s[n][r] - mnew); s[n][r] = e; rs += e; }
      rs += __shfl_xor(rs, 1, 16);
      rs += __shfl_xor(rs, 2, 16);
      rs += __shfl_xor(rs, 4, 16);
      rs += __shfl_xor(rs, 8, 16);
      l[r] = l[r] * alpha[r] + rs;
#pragma unroll
      for (int n = 0; n < 4; ++n) o[n][r] *= alpha[r];
    }

    // route P through per-wave LDS to re-fragment (C-layout -> A-layout)
    {
      const int cc = lane & 15, rb = (lane >> 4) * 8;
#pragma unroll
      for (int n = 0; n < 4; ++n)
#pragma unroll
        for (int r = 0; r < 8; ++r)
          sP[wave][(rb + r) * 64 + n * 16 + cc] = (__bf16)s[n][r];
    }
    // O += P V   (per-wave LDS RAW: hardware DS ordering + compiler waits)
#pragma unroll
    for (int kk = 0; kk < 2; ++kk) {
      v16bf pf = load_frag16(&sP[wave][0] + kk * 32, 64, lane);
#pragma unroll
      for (int n = 0; n < 4; ++n) {
        v16bf vf = load_frag16_kmajor(sV + (size_t)(kk * 32) * 64 + n * 16, 64, lane);
        o[n] = WMMA_BF16(pf, vf, o[n]);
      }
    }
    __syncthreads();
  }

  const int cc = lane & 15, rb = (lane >> 4) * 8;
#pragma unroll
  for (int n = 0; n < 4; ++n)
#pragma unroll
    for (int r = 0; r < 8; ++r) {
      size_t row = (size_t)b * GT + q0 + wave * 16 + rb + r;
      ctx[row * GD + h * GDH + n * 16 + cc] = (__bf16)(o[n][r] / l[r]);
    }
}

// ---------------------------------------------------------------------------
// TDM / fallback weight preload into LDS
// ---------------------------------------------------------------------------
#if HAVE_TDM
__device__ inline void tdm_load_2d(unsigned lds_off, const void* gaddr,
                                   unsigned rowElems, unsigned nRows,
                                   unsigned strideElems, unsigned tensorRows) {
  unsigned long long ga = (unsigned long long)(size_t)gaddr;
  u32x4 g0;
  g0[0] = 1u;                                         // count=1 user descriptor
  g0[1] = lds_off;                                    // lds_addr (bytes)
  g0[2] = (unsigned)(ga & 0xFFFFFFFFu);               // global_addr[31:0]
  g0[3] = (unsigned)((ga >> 32) & 0x01FFFFFFu) | 0x80000000u; // addr[56:32] | type=2
  i32x8 g1;
  g1[0] = (int)(1u << 16);                            // data_size=1 (2B), wg_mask=0
  g1[1] = (int)((rowElems & 0xFFFFu) << 16);          // tensor_dim0[15:0]
  g1[2] = (int)(((rowElems >> 16) & 0xFFFFu) | ((tensorRows & 0xFFFFu) << 16));
  g1[3] = (int)(((tensorRows >> 16) & 0xFFFFu) | ((rowElems & 0xFFFFu) << 16)); // tile_dim0
  g1[4] = (int)(nRows & 0xFFFFu);                     // tile_dim1 (tile_dim2=0)
  g1[5] = (int)strideElems;                           // tensor_dim0_stride[31:0]
  g1[6] = 0;
  g1[7] = 0;
  i32x4 z4 = {0, 0, 0, 0};                            // groups 2/3 unused (2D tensor)
  i32x8 z8 = {0, 0, 0, 0, 0, 0, 0, 0};                // extra group (clang-23 form)
  __builtin_amdgcn_tensor_load_to_lds(g0, g1, z4, z4, z8, 0);
}
#endif

__device__ inline void lds_copy_tile(__bf16* dst, const __bf16* src,
                                     int rowElems, int nRows, int strideElems) {
  const int chunksPerRow = rowElems >> 3;
  const int total = chunksPerRow * nRows;
  for (int i = threadIdx.x; i < total; i += blockDim.x) {
    const int r = i / chunksPerRow, cc = i - r * chunksPerRow;
    *(uint4*)(dst + (size_t)i * 8) =
        *(const uint4*)(src + (size_t)r * strideElems + (size_t)cc * 8);
  }
}

// generation-counter grid barrier (bar[0]=count, bar[1]=generation)
__device__ inline void gridbar(unsigned* bar, unsigned nwg) {
  __syncthreads();
  if (threadIdx.x == 0) {
    volatile unsigned* gen = bar + 1;
    unsigned g = *gen;
    __threadfence();
    unsigned old = atomicAdd(bar, 1u);
    if (old == nwg - 1u) {
      *(volatile unsigned*)bar = 0u;
      __threadfence();
      atomicAdd((unsigned*)(bar + 1), 1u);
    } else {
      while (*gen == g) __builtin_amdgcn_s_sleep(2);
    }
  }
  __syncthreads();
}

// ---------------------------------------------------------------------------
// Persistent recurrent scan. 64 WGs x 128 threads (4 waves). WG w owns h
// columns [16w, 16w+16). Weight slices (w_hh: 3x16x1024, gate_w: 16x2048,
// bf16 = 160KB) live in LDS for the whole scan (TDM-preloaded). h (8x1024)
// ping-pongs through L2 between grid barriers (2 per substep).
// ---------------------------------------------------------------------------
__global__ __launch_bounds__(SCAN_THREADS)
void scan_kernel(const float* __restrict__ gi,      // [B*T, 3D] f32 (b_ih folded in)
                 const float* __restrict__ a_f,     // [B*T, D]  f32
                 const __bf16* __restrict__ a_b,    // [B*T, D]  bf16
                 const __bf16* __restrict__ whh,    // [3D, D]   bf16
                 const float* __restrict__ b_hh,    // [3D]
                 const __bf16* __restrict__ wgate,  // [D, 2D]   bf16
                 const float* __restrict__ gate_b,  // [D]
                 __bf16* __restrict__ h16,          // [16, D] bf16 (rows 8..15 = 0)
                 float* __restrict__ out,           // [B, T, D]
                 unsigned* __restrict__ bar) {
  extern __shared__ char smem[];
  __bf16* lw   = (__bf16*)smem;            // 3 * 16*1024
  __bf16* lgw  = lw + 3 * 16 * 1024;       // 16*2048
  __bf16* lh   = lgw + 16 * 2048;          // 16*1024
  __bf16* la   = lh + 16 * 1024;           // 16*1024
  float*  sG   = (float*)(la + 16 * 1024); // 3*16*16
  float*  sGP  = sG + 3 * 256;             // 4*16*16
  float*  sGPa = sGP + 4 * 256;            // 16*16

  const int tid = threadIdx.x, lane = tid & 31, wave = tid >> 5;
  const int c0 = blockIdx.x * 16;

  // ---- one-time weight preload into LDS ----
#if HAVE_TDM
  if (tid == 0) {
    for (int g = 0; g < 3; ++g)
      tdm_load_2d((unsigned)((char*)(lw + (size_t)g * 16 * 1024) - smem),
                  whh + (size_t)(g * GD + c0) * GD, GD, 16, GD, 3 * GD);
    tdm_load_2d((unsigned)((char*)lgw - smem),
                wgate + (size_t)c0 * 2 * GD, 2 * GD, 16, 2 * GD, GD);
    __builtin_amdgcn_s_wait_tensorcnt(0);
  }
#else
  for (int g = 0; g < 3; ++g)
    lds_copy_tile(lw + (size_t)g * 16 * 1024, whh + (size_t)(g * GD + c0) * GD, GD, 16, GD);
  lds_copy_tile(lgw, wgate + (size_t)c0 * 2 * GD, 2 * GD, 16, 2 * GD);
#endif
  // zero M-padding rows 8..15 of the h / a staging tiles
  for (int i = tid; i < 8 * GD; i += blockDim.x) {
    lh[8 * GD + i] = (__bf16)0.f;
    la[8 * GD + i] = (__bf16)0.f;
  }
  __syncthreads();

  const int b = tid >> 4, c = tid & 15;     // 128 threads <-> 8 batch x 16 cols
  const float bhh_r = b_hh[c0 + c];
  const float bhh_z = b_hh[GD + c0 + c];
  const float bhh_n = b_hh[2 * GD + c0 + c];
  const float gbv   = gate_b[c0 + c];
  float hcur = 0.f;

  for (int t = 0; t < GT; ++t) {
    // stage a_t (8 rows x 1024) into la
    {
      const int r = tid >> 4, seg = tid & 15;
      const __bf16* src = a_b + ((size_t)r * GT + t) * GD + seg * 64;
      __bf16* dst = la + r * GD + seg * 64;
#pragma unroll
      for (int q = 0; q < 8; ++q)
        *(uint4*)(dst + q * 8) = *(const uint4*)(src + q * 8);
    }
    __syncthreads();

    // wave 3: a-half of gate GEMM (constant across the 4 substeps)
    if (wave == 3) {
      v8f acc = {};
      for (int k0 = 0; k0 < GD; k0 += 32) {
        v16bf af = load_frag16(la + k0, GD, lane);
        v16bf bf = load_frag16(lgw + GD + k0, 2 * GD, lane);
        acc = WMMA_BF16(af, bf, acc);
      }
      store_c_tile(sGPa, 16, lane, acc);
    }

    const size_t rowBT = (size_t)b * GT + t;
    const float gir = gi[rowBT * 3 * GD + c0 + c];
    const float giz = gi[rowBT * 3 * GD + GD + c0 + c];
    const float gin = gi[rowBT * 3 * GD + 2 * GD + c0 + c];
    const float av  = a_f[rowBT * GD + c0 + c];
    float hgru = 0.f;

    for (int s = 0; s < NSTEPS; ++s) {
      // stage current full h (8x1024) into lh
      {
        const int r = tid >> 4, seg = tid & 15;
        const __bf16* src = h16 + r * GD + seg * 64;
        __bf16* dst = lh + r * GD + seg * 64;
#pragma unroll
        for (int q = 0; q < 8; ++q)
          *(uint4*)(dst + q * 8) = *(const uint4*)(src + q * 8);
      }
      __syncthreads();

      // gh[:, owned cols] for gates r/z/n -- waves 0..2, full K=1024
      if (wave < 3) {
        v8f acc = {};
        const __bf16* ws = lw + (size_t)wave * 16 * 1024;
        for (int k0 = 0; k0 < GD; k0 += 32) {
          v16bf af = load_frag16(lh + k0, GD, lane);
          v16bf bf = load_frag16(ws + k0, GD, lane);
          acc = WMMA_BF16(af, bf, acc);
        }
        store_c_tile(sG + wave * 256, 16, lane, acc);
      }
      __syncthreads();

      // GRU elementwise update for (b, c)
      {
        float ghr = sG[0 * 256 + b * 16 + c] + bhh_r;
        float ghz = sG[1 * 256 + b * 16 + c] + bhh_z;
        float ghn = sG[2 * 256 + b * 16 + c] + bhh_n;
        float r_ = 1.f / (1.f + __expf(-(gir + ghr)));
        float z_ = 1.f / (1.f + __expf(-(giz + ghz)));
        float n_ = tanhf(gin + r_ * ghn);
        hgru = (1.f - z_) * n_ + z_ * hcur;
        h16[b * GD + c0 + c] = (__bf16)hgru;
      }
      __threadfence();
      gridbar(bar, SCAN_NWG);

      // stage updated h
      {
        const int r = tid >> 4, seg = tid & 15;
        const __bf16* src = h16 + r * GD + seg * 64;
        __bf16* dst = lh + r * GD + seg * 64;
#pragma unroll
        for (int q = 0; q < 8; ++q)
          *(uint4*)(dst + q * 8) = *(const uint4*)(src + q * 8);
      }
      __syncthreads();

      // gate GEMM h-half: 4 waves split K=1024 into 256-chunks
      {
        v8f acc = {};
        const int kb = wave * 256;
        for (int k0 = 0; k0 < 256; k0 += 32) {
          v16bf af = load_frag16(lh + kb + k0, GD, lane);
          v16bf bf = load_frag16(lgw + kb + k0, 2 * GD, lane);
          acc = WMMA_BF16(af, bf, acc);
        }
        store_c_tile(sGP + wave * 256, 16, lane, acc);
      }
      __syncthreads();

      {
        float gsum = sGP[0 + b * 16 + c] + sGP[256 + b * 16 + c] +
                     sGP[512 + b * 16 + c] + sGP[768 + b * 16 + c] +
                     sGPa[b * 16 + c] + gbv;
        float g_ = 1.f / (1.f + __expf(-gsum));
        float hn = g_ * hgru + (1.f - g_) * av;
        hcur = hn;
        h16[b * GD + c0 + c] = (__bf16)hn;
        if (s == NSTEPS - 1) out[rowBT * GD + c0 + c] = hn;
      }
      __threadfence();
      gridbar(bar, SCAN_NWG);
    }
  }
}

// ---------------------------------------------------------------------------
// Host launch
// ---------------------------------------------------------------------------
extern "C" void kernel_launch(void* const* d_in, const int* in_sizes, int n_in,
                              void* d_out, int out_size, void* d_ws, size_t ws_size,
                              hipStream_t stream) {
  const float* x       = (const float*)d_in[0];
  const float* ipw     = (const float*)d_in[1];
  const float* ipb     = (const float*)d_in[2];
  const float* opw     = (const float*)d_in[3];
  const float* opb     = (const float*)d_in[4];
  const float* wih     = (const float*)d_in[5];
  const float* whh     = (const float*)d_in[6];
  const float* bih     = (const float*)d_in[7];
  const float* bhh     = (const float*)d_in[8];
  const float* gatew   = (const float*)d_in[9];
  const float* gateb   = (const float*)d_in[10];
  float* out = (float*)d_out;

  const size_t BT = (size_t)GB * GT;
  char* p = (char*)d_ws;
  auto alloc = [&](size_t bytes) -> char* {
    char* r = p;
    p += (bytes + 255) & ~(size_t)255;
    return r;
  };
  __bf16* xb    = (__bf16*)alloc(BT * GD * 2);             // 16 MB
  __bf16* wqkvb = (__bf16*)alloc((size_t)3 * GD * GD * 2); // 6 MB
  __bf16* woutb = (__bf16*)alloc((size_t)GD * GD * 2);     // 2 MB
  __bf16* wihb  = (__bf16*)alloc((size_t)3 * GD * GD * 2); // 6 MB
  __bf16* whhb  = (__bf16*)alloc((size_t)3 * GD * GD * 2); // 6 MB
  __bf16* wgtb  = (__bf16*)alloc((size_t)GD * 2 * GD * 2); // 4 MB
  __bf16* qkvb  = (__bf16*)alloc(BT * 3 * GD * 2);         // 48 MB
  __bf16* ctxb  = (__bf16*)alloc(BT * GD * 2);             // 16 MB
  float*  a_f   = (float*)alloc(BT * GD * 4);              // 32 MB
  __bf16* a_b   = (__bf16*)alloc(BT * GD * 2);             // 16 MB
  float*  gibuf = (float*)alloc(BT * 3 * GD * 4);          // 96 MB
  __bf16* h16   = (__bf16*)alloc((size_t)16 * GD * 2);     // 32 KB
  unsigned* bar = (unsigned*)alloc(256);

  auto cast = [&](const float* s, __bf16* d, int n) {
    cast_f32_bf16<<<(n + 255) / 256, 256, 0, stream>>>(s, d, n);
  };
  cast(x, xb, (int)(BT * GD));
  cast(ipw, wqkvb, 3 * GD * GD);
  cast(opw, woutb, GD * GD);
  cast(wih, wihb, 3 * GD * GD);
  cast(whh, whhb, 3 * GD * GD);
  cast(gatew, wgtb, GD * 2 * GD);

  init_state<<<(16 * GD + 255) / 256, 256, 0, stream>>>(h16, bar);

  // qkv = x @ in_proj_w^T + in_proj_b   (bf16 out for attention)
  gemm_bf16_nt<<<dim3((unsigned)(BT / 128), 3 * GD / 128), 256, 0, stream>>>(
      xb, wqkvb, ipb, nullptr, qkvb, (int)BT, 3 * GD, GD);
  // gi = x @ w_ih^T + b_ih              (f32 out, consumed by the scan)
  gemm_bf16_nt<<<dim3((unsigned)(BT / 128), 3 * GD / 128), 256, 0, stream>>>(
      xb, wihb, bih, gibuf, nullptr, (int)BT, 3 * GD, GD);
  // attention
  flash_attn<<<dim3(GB * GH, GT / 64), 128, 0, stream>>>(qkvb, ctxb);
  // attn_out = ctx @ out_proj_w^T + out_proj_b   (f32 + bf16)
  gemm_bf16_nt<<<dim3((unsigned)(BT / 128), GD / 128), 256, 0, stream>>>(
      ctxb, woutb, opb, a_f, a_b, (int)BT, GD, GD);

  // persistent recurrent scan (dynamic LDS = 232 KB, fits in 320 KB/WGP)
  const size_t scan_lds =
      (size_t)(3 * 16 * 1024 + 16 * 2048 + 16 * 1024 + 16 * 1024) * 2 +
      (size_t)(3 * 256 + 4 * 256 + 256) * 4;
  (void)hipFuncSetAttribute((const void*)scan_kernel,
                            hipFuncAttributeMaxDynamicSharedMemorySize,
                            (int)scan_lds);
  scan_kernel<<<SCAN_NWG, SCAN_THREADS, scan_lds, stream>>>(
      gibuf, a_f, a_b, whhb, bhh, wgtb, gateb, h16, out, bar);

  (void)in_sizes; (void)n_in; (void)out_size; (void)ws_size;
}
